// MiniMaxM2SparseMoeBlock_88579405513102
// MI455X (gfx1250) — compile-verified, matrix operand order checked by reading, not run on hardware
//
#include <hip/hip_runtime.h>
#include <hip/hip_bf16.h>
#include <math.h>

// ---------------------------------------------------------------------------
// MoE (MiniMax-M2 style sparse block) for gfx1250 / MI455X.
// top-2 compaction + bf16 WMMA GEMMs.
// Staging: A tiles (gathered rows) via global_load_async_to_lds_b128
// (ASYNCcnt), B tiles (dense weights) via TDM tensor_load_to_lds (TENSORcnt),
// double-buffered LDS. Workspace requirement: ~236 MB (WS_NEEDED).
// ---------------------------------------------------------------------------

#define TOKENS 16384   // B*S
#define HDIM   2048
#define IDIM   1024
#define NEXP   8

#define BM 128
#define BN 64
#define BK 64
#define LDK (BK + 8)          // 72 bf16 = 144B row stride: 16B-aligned, bank-safe
#define ROWB (LDK * 2)        // 144 bytes
#define A_BYTES (BM * ROWB)   // 18432
#define B_BYTES (BN * ROWB)   // 9216
#define STAGE_A (A_BYTES + 2 * B_BYTES)  // 36864 (A + B1 + B3)
#define STAGE_B (A_BYTES + B_BYTES)      // 27648 (A + B)

typedef __attribute__((ext_vector_type(16))) __bf16 bf16x16;
typedef __attribute__((ext_vector_type(8)))  __bf16 bf16x8;
typedef __attribute__((ext_vector_type(4)))  __bf16 bf16x4;
typedef __attribute__((ext_vector_type(8)))  float  f32x8;
typedef __attribute__((ext_vector_type(4)))  unsigned u32x4;
typedef __attribute__((ext_vector_type(8)))  int      i32x8;

union Frag { bf16x16 v; bf16x8 h[2]; };

__device__ inline f32x8 wmma_bf16(const Frag& a, const Frag& b, f32x8 c) {
    return __builtin_amdgcn_wmma_f32_16x16x32_bf16(false, a.v, false, b.v,
                                                   (short)0, c, false, false);
}

// Per-lane async global->LDS (ASYNCcnt), GV mode.
__device__ inline void async_b128(unsigned lds, unsigned long long gaddr) {
    asm volatile("global_load_async_to_lds_b128 %0, %1, off"
                 :: "v"(lds), "v"(gaddr) : "memory");
}
__device__ inline void wait_async4() { asm volatile("s_wait_asynccnt 0x4" ::: "memory"); }
__device__ inline void wait_async0() { asm volatile("s_wait_asynccnt 0x0" ::: "memory"); }

// TDM 2-D tile load (TENSORcnt). D# group1 is constant per kernel; group0
// carries {count=1, lds_addr, global_addr, type=2}.  ISA 08_async_tensor §8.
// group1 word0: data_size=1(2B) [17:16], pad_enable [20],
//               pad_interval=4 (32 DWORDs = 128B row) [24:22],
//               pad_amount=3 (4 DWORDs = 16B pad)     [31:25]
//   => rows land at 144B stride, matching ROWB.
#define TDM_G1_WORD0 ((int)((1u << 16) | (1u << 20) | (4u << 22) | (3u << 25)))
__device__ inline i32x8 tdm_desc1(int dim0_stride) {
    i32x8 g1;
    g1[0] = TDM_G1_WORD0;
    g1[1] = (int)(64u << 16);   // tensor_dim0[15:0]=64 at bits[79:64->word1 hi]
    g1[2] = (int)(64u << 16);   // tensor_dim0 hi=0 | tensor_dim1[15:0]=64
    g1[3] = (int)(64u << 16);   // tensor_dim1 hi=0 | tile_dim0=64
    g1[4] = 64;                 // tile_dim1=64, tile_dim2=0
    g1[5] = dim0_stride;        // tensor_dim0_stride low 32 (elements)
    g1[6] = 0;                  // stride hi | tensor_dim1_stride lo (unused 2D)
    g1[7] = 0;
    return g1;
}
__device__ inline void tdm_load_2d(unsigned lds, unsigned long long gaddr,
                                   i32x8 g1) {
    u32x4 g0;
    g0[0] = 1u;                                     // count=1, user mode
    g0[1] = lds;                                    // lds_addr
    g0[2] = (unsigned)gaddr;                        // global_addr[31:0]
    g0[3] = (unsigned)((gaddr >> 32) & 0x01FFFFFFu) // global_addr[56:32]
            | (2u << 30);                           // type=2 ("image")
    asm volatile("tensor_load_to_lds %0, %1" :: "s"(g0), "s"(g1) : "memory");
}

__device__ inline unsigned lds_addr32(const void* p) {
    // LDS aperture: flat addr[31:0] is the wave-relative LDS byte address.
    return (unsigned)(unsigned long long)(size_t)p;
}

// ---------------------------------------------------------------------------
// fp32 -> bf16 conversion
// ---------------------------------------------------------------------------
__global__ void cvt_f32_to_bf16(const float* __restrict__ src,
                                __bf16* __restrict__ dst, size_t n4) {
    size_t i = (size_t)blockIdx.x * blockDim.x + threadIdx.x;
    size_t stride = (size_t)gridDim.x * blockDim.x;
    for (; i < n4; i += stride) {
        float4 v = ((const float4*)src)[i];
        bf16x4 o = { (__bf16)v.x, (__bf16)v.y, (__bf16)v.z, (__bf16)v.w };
        ((bf16x4*)dst)[i] = o;
    }
}

// ---------------------------------------------------------------------------
// Router: one wave32 per token; top-2; bucket append.
// ---------------------------------------------------------------------------
__global__ void router_kernel(const float* __restrict__ x,
                              const float* __restrict__ gw,
                              float* __restrict__ logits_out,
                              unsigned* __restrict__ cnt,
                              unsigned* __restrict__ idxbuf,
                              float* __restrict__ wgtbuf) {
    int t = blockIdx.x * (blockDim.x >> 5) + (threadIdx.x >> 5);
    int lane = threadIdx.x & 31;
    if (t >= TOKENS) return;

    float acc[NEXP];
#pragma unroll
    for (int e = 0; e < NEXP; ++e) acc[e] = 0.f;

    const float* xr = x + (size_t)t * HDIM;
    for (int h = lane; h < HDIM; h += 32) {
        float xv = xr[h];
#pragma unroll
        for (int e = 0; e < NEXP; ++e) acc[e] += xv * gw[e * HDIM + h];
    }
#pragma unroll
    for (int e = 0; e < NEXP; ++e) {
#pragma unroll
        for (int off = 16; off; off >>= 1) acc[e] += __shfl_xor(acc[e], off, 32);
    }

    if (lane == 0) {
#pragma unroll
        for (int e = 0; e < NEXP; ++e) logits_out[(size_t)t * NEXP + e] = acc[e];
        int e0 = 0;
#pragma unroll
        for (int e = 1; e < NEXP; ++e) if (acc[e] > acc[e0]) e0 = e;
        int e1 = (e0 == 0) ? 1 : 0;
#pragma unroll
        for (int e = 0; e < NEXP; ++e)
            if (e != e0 && acc[e] > acc[e1]) e1 = e;
        float mx = fmaxf(acc[e0], acc[e1]);
        float p0 = __expf(acc[e0] - mx);
        float p1 = __expf(acc[e1] - mx);
        float inv = 1.0f / fmaxf(p0 + p1, 1e-12f);
        unsigned s0 = atomicAdd(&cnt[e0], 1u);
        idxbuf[(size_t)e0 * TOKENS + s0] = (unsigned)t;
        wgtbuf[(size_t)e0 * TOKENS + s0] = p0 * inv;
        unsigned s1 = atomicAdd(&cnt[e1], 1u);
        idxbuf[(size_t)e1 * TOKENS + s1] = (unsigned)t;
        wgtbuf[(size_t)e1 * TOKENS + s1] = p1 * inv;
    }
}

__global__ void prefix_kernel(const unsigned* __restrict__ cnt,
                              unsigned* __restrict__ offs) {
    if (threadIdx.x == 0) {
        unsigned s = 0;
        for (int e = 0; e < NEXP; ++e) { offs[e] = s; s += cnt[e]; }
    }
}

// ---------------------------------------------------------------------------
// Phase A: act = silu(X_e . W1[e]^T) * (X_e . W3[e]^T).
// ---------------------------------------------------------------------------
__global__ __launch_bounds__(256)
void moe_swiglu_kernel(const __bf16* __restrict__ xbf,
                       const __bf16* __restrict__ w1bf,
                       const __bf16* __restrict__ w3bf,
                       const unsigned* __restrict__ cnt,
                       const unsigned* __restrict__ offs,
                       const unsigned* __restrict__ idxbuf,
                       __bf16* __restrict__ act) {
    const int e    = blockIdx.z;
    const int row0 = blockIdx.y * BM;
    const int n0   = blockIdx.x * BN;
    const int ne   = (int)cnt[e];
    if (row0 >= ne) return;

    __shared__ __align__(16) char smem[2 * STAGE_A];
    __shared__ unsigned sTok[BM];

    const int tid = threadIdx.x;
    if (tid < BM) {
        int r = row0 + tid;
        if (r >= ne) r = ne - 1;          // clamp; epilogue masks invalid rows
        sTok[tid] = idxbuf[(size_t)e * TOKENS + r];
    }
    __syncthreads();

    const unsigned smem0 = lds_addr32(smem);
    unsigned long long gA[4];
    unsigned lAo[4];
#pragma unroll
    for (int it = 0; it < 4; ++it) {
        int c = tid + it * 256;
        int row = c >> 3, c8 = (c & 7) * 8;
        gA[it]  = (unsigned long long)(size_t)(xbf + (size_t)sTok[row] * HDIM + c8);
        lAo[it] = smem0 + row * ROWB + c8 * 2;
    }
    const unsigned long long gB1base =
        (unsigned long long)(size_t)(w1bf + ((size_t)e * IDIM + n0) * HDIM);
    const unsigned long long gB3base =
        (unsigned long long)(size_t)(w3bf + ((size_t)e * IDIM + n0) * HDIM);
    const i32x8 g1 = tdm_desc1(HDIM);

    const int lane = tid & 31, wave = tid >> 5;
    const int wm = wave & 3, wn = wave >> 2;
    const int lrow = lane & 15, khalf = lane >> 4;

    auto issue = [&](int buf, int kk) {
        unsigned boff = (unsigned)buf * STAGE_A;
        unsigned long long koff = (unsigned long long)kk * 2;
#pragma unroll
        for (int it = 0; it < 4; ++it) async_b128(lAo[it] + boff, gA[it] + koff);
        if (wave == 0) {   // TDM tile loads: one per 64x64 weight tile
            tdm_load_2d(smem0 + boff + A_BYTES,           gB1base + koff, g1);
            tdm_load_2d(smem0 + boff + A_BYTES + B_BYTES, gB3base + koff, g1);
        }
    };

    f32x8 accG[2][2] = {{{0.f}}};
    f32x8 accU[2][2] = {{{0.f}}};

    auto compute = [&](const char* sb) {
#pragma unroll
        for (int ks = 0; ks < BK; ks += 32) {
            Frag a[2], b1[2], b3[2];
#pragma unroll
            for (int m2 = 0; m2 < 2; ++m2) {
                const char* p = sb + (wm * 32 + m2 * 16 + lrow) * ROWB
                                   + (ks + 8 * khalf) * 2;
                a[m2].h[0] = *(const bf16x8*)p;
                a[m2].h[1] = *(const bf16x8*)(p + 32);
            }
#pragma unroll
            for (int n2 = 0; n2 < 2; ++n2) {
                const char* p1 = sb + A_BYTES + (wn * 32 + n2 * 16 + lrow) * ROWB
                                    + (ks + 8 * khalf) * 2;
                b1[n2].h[0] = *(const bf16x8*)p1;
                b1[n2].h[1] = *(const bf16x8*)(p1 + 32);
                const char* p3 = p1 + B_BYTES;
                b3[n2].h[0] = *(const bf16x8*)p3;
                b3[n2].h[1] = *(const bf16x8*)(p3 + 32);
            }
#pragma unroll
            for (int m2 = 0; m2 < 2; ++m2)
#pragma unroll
                for (int n2 = 0; n2 < 2; ++n2) {
                    accG[m2][n2] = wmma_bf16(a[m2], b1[n2], accG[m2][n2]);
                    accU[m2][n2] = wmma_bf16(a[m2], b3[n2], accU[m2][n2]);
                }
        }
    };

    issue(0, 0);
    int buf = 0;
    for (int kk = BK; kk < HDIM; kk += BK) {        // steady state, branch-free
        issue(buf ^ 1, kk);
        wait_async4();
        if (wave == 0) __builtin_amdgcn_s_wait_tensorcnt(2);
        __syncthreads();
        compute(smem + buf * STAGE_A);
        __syncthreads();
        buf ^= 1;
    }
    wait_async0();
    if (wave == 0) __builtin_amdgcn_s_wait_tensorcnt(0);
    __syncthreads();
    compute(smem + buf * STAGE_A);                  // peeled last stage

    const size_t slotbase = (size_t)offs[e] + (size_t)row0;
#pragma unroll
    for (int m2 = 0; m2 < 2; ++m2)
#pragma unroll
        for (int n2 = 0; n2 < 2; ++n2)
#pragma unroll
            for (int r = 0; r < 8; ++r) {
                int rloc = wm * 32 + m2 * 16 + r + 8 * khalf;
                if (row0 + rloc < ne) {
                    int col = n0 + wn * 32 + n2 * 16 + lrow;
                    float g = accG[m2][n2][r];
                    float u = accU[m2][n2][r];
                    float val = (g / (1.0f + __expf(-g))) * u;
                    act[(slotbase + rloc) * IDIM + col] = (__bf16)val;
                }
            }
}

// ---------------------------------------------------------------------------
// Phase B: y = act . W2[e]^T ; scale by combine weight; atomic add into out.
// ---------------------------------------------------------------------------
__global__ __launch_bounds__(256)
void moe_down_kernel(const __bf16* __restrict__ act,
                     const __bf16* __restrict__ w2bf,
                     const unsigned* __restrict__ cnt,
                     const unsigned* __restrict__ offs,
                     const unsigned* __restrict__ idxbuf,
                     const float* __restrict__ wgtbuf,
                     float* __restrict__ out) {
    const int e    = blockIdx.z;
    const int row0 = blockIdx.y * BM;
    const int n0   = blockIdx.x * BN;
    const int ne   = (int)cnt[e];
    if (row0 >= ne) return;
    const size_t slot0 = (size_t)offs[e];

    __shared__ __align__(16) char smem[2 * STAGE_B];

    const int tid = threadIdx.x;
    const unsigned smem0 = lds_addr32(smem);
    unsigned long long gA[4];
    unsigned lAo[4];
#pragma unroll
    for (int it = 0; it < 4; ++it) {
        int c = tid + it * 256;
        int row = c >> 3, c8 = (c & 7) * 8;
        int r = row0 + row; if (r >= ne) r = ne - 1;   // clamp
        gA[it]  = (unsigned long long)(size_t)(act + (slot0 + r) * IDIM + c8);
        lAo[it] = smem0 + row * ROWB + c8 * 2;
    }
    const unsigned long long gBbase =
        (unsigned long long)(size_t)(w2bf + ((size_t)e * HDIM + n0) * IDIM);
    const i32x8 g1 = tdm_desc1(IDIM);

    const int lane = tid & 31, wave = tid >> 5;
    const int wm = wave & 3, wn = wave >> 2;
    const int lrow = lane & 15, khalf = lane >> 4;

    auto issue = [&](int buf, int kk) {
        unsigned boff = (unsigned)buf * STAGE_B;
        unsigned long long koff = (unsigned long long)kk * 2;
#pragma unroll
        for (int it = 0; it < 4; ++it) async_b128(lAo[it] + boff, gA[it] + koff);
        if (wave == 0)
            tdm_load_2d(smem0 + boff + A_BYTES, gBbase + koff, g1);
    };

    f32x8 acc[2][2] = {{{0.f}}};

    auto compute = [&](const char* sb) {
#pragma unroll
        for (int ks = 0; ks < BK; ks += 32) {
            Frag a[2], b[2];
#pragma unroll
            for (int m2 = 0; m2 < 2; ++m2) {
                const char* p = sb + (wm * 32 + m2 * 16 + lrow) * ROWB
                                   + (ks + 8 * khalf) * 2;
                a[m2].h[0] = *(const bf16x8*)p;
                a[m2].h[1] = *(const bf16x8*)(p + 32);
            }
#pragma unroll
            for (int n2 = 0; n2 < 2; ++n2) {
                const char* p = sb + A_BYTES + (wn * 32 + n2 * 16 + lrow) * ROWB
                                   + (ks + 8 * khalf) * 2;
                b[n2].h[0] = *(const bf16x8*)p;
                b[n2].h[1] = *(const bf16x8*)(p + 32);
            }
#pragma unroll
            for (int m2 = 0; m2 < 2; ++m2)
#pragma unroll
                for (int n2 = 0; n2 < 2; ++n2)
                    acc[m2][n2] = wmma_bf16(a[m2], b[n2], acc[m2][n2]);
        }
    };

    issue(0, 0);
    int buf = 0;
    for (int kk = BK; kk < IDIM; kk += BK) {
        issue(buf ^ 1, kk);
        wait_async4();
        if (wave == 0) __builtin_amdgcn_s_wait_tensorcnt(1);
        __syncthreads();
        compute(smem + buf * STAGE_B);
        __syncthreads();
        buf ^= 1;
    }
    wait_async0();
    if (wave == 0) __builtin_amdgcn_s_wait_tensorcnt(0);
    __syncthreads();
    compute(smem + buf * STAGE_B);

#pragma unroll
    for (int m2 = 0; m2 < 2; ++m2)
#pragma unroll
        for (int n2 = 0; n2 < 2; ++n2)
#pragma unroll
            for (int r = 0; r < 8; ++r) {
                int rloc = wm * 32 + m2 * 16 + r + 8 * khalf;
                if (row0 + rloc < ne) {
                    unsigned tok = idxbuf[(size_t)e * TOKENS + row0 + rloc];
                    float w = wgtbuf[(size_t)e * TOKENS + row0 + rloc];
                    int col = n0 + wn * 32 + n2 * 16 + lrow;
                    atomicAdd(&out[(size_t)tok * HDIM + col], acc[m2][n2][r] * w);
                }
            }
}

// ---------------------------------------------------------------------------
// Host launch
// ---------------------------------------------------------------------------
static constexpr size_t T_  = TOKENS, H_ = HDIM, I_ = IDIM, E_ = NEXP;
static constexpr size_t CNT_OFF  = 0;
static constexpr size_t OFFS_OFF = 256;
static constexpr size_t IDX_OFF  = 1024;
static constexpr size_t WGT_OFF  = IDX_OFF + E_ * T_ * 4;
static constexpr size_t XBF_OFF  = (WGT_OFF + E_ * T_ * 4 + 255) & ~(size_t)255;
static constexpr size_t W1_OFF   = XBF_OFF + T_ * H_ * 2;
static constexpr size_t W3_OFF   = W1_OFF + E_ * I_ * H_ * 2;
static constexpr size_t W2_OFF   = W3_OFF + E_ * I_ * H_ * 2;
static constexpr size_t ACT_OFF  = W2_OFF + E_ * H_ * I_ * 2;
static constexpr size_t WS_NEEDED = ACT_OFF + 2 * T_ * I_ * 2;   // ~236 MB

extern "C" void kernel_launch(void* const* d_in, const int* in_sizes, int n_in,
                              void* d_out, int out_size, void* d_ws, size_t ws_size,
                              hipStream_t stream) {
    const float* x  = (const float*)d_in[0];
    const float* gw = (const float*)d_in[1];
    const float* w1 = (const float*)d_in[2];
    const float* w2 = (const float*)d_in[3];
    const float* w3 = (const float*)d_in[4];
    float* out = (float*)d_out;

    char* ws = (char*)d_ws;
    unsigned* cnt    = (unsigned*)(ws + CNT_OFF);
    unsigned* offs   = (unsigned*)(ws + OFFS_OFF);
    unsigned* idxbuf = (unsigned*)(ws + IDX_OFF);
    float*    wgtbuf = (float*)(ws + WGT_OFF);
    __bf16*   xbf    = (__bf16*)(ws + XBF_OFF);
    __bf16*   w1bf   = (__bf16*)(ws + W1_OFF);
    __bf16*   w3bf   = (__bf16*)(ws + W3_OFF);
    __bf16*   w2bf   = (__bf16*)(ws + W2_OFF);
    __bf16*   actbuf = (__bf16*)(ws + ACT_OFF);

    hipMemsetAsync(cnt, 0, 256, stream);
    hipMemsetAsync(out, 0, T_ * H_ * sizeof(float), stream);

    auto cvt_blocks = [](size_t n4) {
        size_t b = (n4 + 255) / 256;
        return (unsigned)(b > 16384 ? 16384 : b);
    };
    size_t nx = T_ * H_ / 4, nw = E_ * I_ * H_ / 4;
    cvt_f32_to_bf16<<<cvt_blocks(nx), 256, 0, stream>>>(x,  xbf,  nx);
    cvt_f32_to_bf16<<<cvt_blocks(nw), 256, 0, stream>>>(w1, w1bf, nw);
    cvt_f32_to_bf16<<<cvt_blocks(nw), 256, 0, stream>>>(w3, w3bf, nw);
    cvt_f32_to_bf16<<<cvt_blocks(nw), 256, 0, stream>>>(w2, w2bf, nw);

    router_kernel<<<TOKENS / 8, 256, 0, stream>>>(x, gw, out + T_ * H_,
                                                  cnt, idxbuf, wgtbuf);
    prefix_kernel<<<1, 32, 0, stream>>>(cnt, offs);

    dim3 gA(IDIM / BN, TOKENS / BM, NEXP);   // (16, 128, 8)
    moe_swiglu_kernel<<<gA, 256, 0, stream>>>(xbf, w1bf, w3bf, cnt, offs,
                                              idxbuf, actbuf);
    dim3 gB(HDIM / BN, TOKENS / BM, NEXP);   // (32, 128, 8)
    moe_down_kernel<<<gB, 256, 0, stream>>>(actbuf, w2bf, cnt, offs,
                                            idxbuf, wgtbuf, out);
    (void)in_sizes; (void)n_in; (void)out_size; (void)ws_size;
}
